// Layer_20976620274275
// MI455X (gfx1250) — matrix operand order repaired and checked by reading, not compiled, not run on hardware
//
#include <hip/hip_runtime.h>
#include <hip/hip_bf16.h>
#include <math.h>

typedef __bf16 bf16;
typedef __attribute__((ext_vector_type(16))) __bf16 v16bf;
typedef __attribute__((ext_vector_type(8)))  float  v8f;
typedef __attribute__((ext_vector_type(4)))  int    v4i;
typedef __attribute__((ext_vector_type(8)))  int    v8i;

#define D_H   1024
#define D_FF  4096
#define BS    8
#define P     2048
#define MROWS (BS * P)          // 16384
#define SEQ   (BS * D_H)        // 8192 scan sequences
#define NCH   16                // scan chunks
#define CLEN  (P / NCH)         // 128 steps per chunk

// ---------------- workspace layout (bytes) ----------------
#define OFF_U    ((size_t)0)                       // u bf16 [16384,4096]  (reused as zf bf16 [16384,4096])
#define OFF_WA   (OFF_U   + (size_t)MROWS*4096*2)  // gate_a_w bf16 [1024,4096]
#define OFF_WB   (OFF_WA  + (size_t)1024*4096*2)
#define OFF_WO   (OFF_WB  + (size_t)1024*4096*2)   // W_o bf16 [1024,1024]
#define OFF_W1   (OFF_WO  + (size_t)1024*1024*2)   // ffn_w1 bf16 [4096,1024]
#define OFF_W2   (OFF_W1  + (size_t)4096*1024*2)   // ffn_w2 bf16 [1024,4096]
#define OFF_ABUF (OFF_W2  + (size_t)1024*4096*2)   // f32 [16384,1024]: a_eff -> h -> out
#define OFF_BBUF (OFF_ABUF+ (size_t)MROWS*1024*4)  // f32 [16384,1024]: b -> res1
#define OFF_Z    (OFF_BBUF+ (size_t)MROWS*1024*4)  // bf16 [16384,1024]: h_bf16 -> z_bf16
#define OFF_CSA  (OFF_Z   + (size_t)MROWS*1024*2)  // f32 [NCH,SEQ]
#define OFF_CSB  (OFF_CSA + (size_t)NCH*SEQ*4)
#define OFF_PRE  (OFF_CSB + (size_t)NCH*SEQ*4)

// ---------------- packing / conversion ----------------

// Build u (first 4096 cols, bf16) from the four concatenated fp32 inputs.
__global__ void pack_u_kernel(const float* __restrict__ x, const float* __restrict__ ypm,
                              const float* __restrict__ ywm, const float* __restrict__ yem,
                              bf16* __restrict__ u) {
    const long total = (long)MROWS * 4096;
    for (long i = (long)blockIdx.x * blockDim.x + threadIdx.x; i < total;
         i += (long)gridDim.x * blockDim.x) {
        long row = i >> 12;
        int  col = (int)(i & 4095);
        int  t   = col >> 10;
        int  w   = col & 1023;
        const float* s = (t == 0) ? x : (t == 1) ? ypm : (t == 2) ? ywm : yem;
        u[i] = (bf16)s[row * 1024 + w];
    }
}

// Strided fp32 -> bf16 (drops trailing columns when srcStride > cols).
__global__ void cvt_f32_bf16_kernel(const float* __restrict__ src, bf16* __restrict__ dst,
                                    long rows, long srcStride, long cols) {
    const long total = rows * cols;
    for (long i = (long)blockIdx.x * blockDim.x + threadIdx.x; i < total;
         i += (long)gridDim.x * blockDim.x) {
        long r = i / cols;
        long c = i - r * cols;
        dst[i] = (bf16)src[r * srcStride + c];
    }
}

// ---------------- WMMA GEMM: C[m,n] = sum_k A[m,k]*B[n,k] (+fused epilogue) ----------------
// A: bf16 [M,K] row-major; B: bf16 [N,K] row-major (weights stored [out,in]).
// MODE: 0=gate_a(sigmoid*carry) 1=gate_b(tanh) 2=W_o(+residual) 3=ffn1(gelu->bf16) 4=ffn2(+residual)

union AFragU { v16bf v; struct { v4i lo, hi; } q; };
union BFragU { v16bf v; v8i q; };

template <int MODE>
__global__ __launch_bounds__(256) void gemm_bf16_wmma(
    const bf16* __restrict__ A, const bf16* __restrict__ Bw,
    float* __restrict__ Cf, bf16* __restrict__ Cb,
    int M, int N, int K,
    const float* __restrict__ bias,
    const float* __restrict__ extraW, int extraStride,
    const float* __restrict__ surprise,
    const float* __restrict__ carry,
    const float* __restrict__ residual)
{
    const int lane = threadIdx.x & 31;
    const int wave = threadIdx.x >> 5;
    const int wm   = wave & 3;     // 4 waves along M
    const int wn   = wave >> 2;    // 2 waves along N
    const int half = lane >> 4;
    const int l15  = lane & 15;

    const int m0 = blockIdx.y * 128 + wm * 32;   // wave tile: 32 (M) x 64 (N)
    const int n0 = blockIdx.x * 128 + wn * 64;

    v8f acc[2][4];
    #pragma unroll
    for (int i = 0; i < 2; ++i)
        #pragma unroll
        for (int j = 0; j < 4; ++j)
            #pragma unroll
            for (int r = 0; r < 8; ++r) acc[i][j][r] = 0.0f;

    // A 16x32 layout (matches B): lane row = l&15; elems e -> K = half*16 + e.
    const bf16* aRow0 = A + (size_t)(m0 + l15) * K;
    const bf16* aRow1 = A + (size_t)(m0 + 16 + l15) * K;
    // B 32x16 layout (same k order as A): lane col = l&15; elems e -> K = half*16 + e.
    const bf16* bRow[4];
    #pragma unroll
    for (int j = 0; j < 4; ++j) bRow[j] = Bw + (size_t)(n0 + 16 * j + l15) * K;

    for (int k0 = 0; k0 < K; k0 += 32) {
        AFragU a0, a1;
        a0.q.lo = *(const v4i*)(aRow0 + k0 + half * 16);
        a0.q.hi = *(const v4i*)(aRow0 + k0 + half * 16 + 8);
        a1.q.lo = *(const v4i*)(aRow1 + k0 + half * 16);
        a1.q.hi = *(const v4i*)(aRow1 + k0 + half * 16 + 8);
        BFragU bf[4];
        #pragma unroll
        for (int j = 0; j < 4; ++j)
            bf[j].q = *(const v8i*)(bRow[j] + k0 + half * 16);

        #pragma unroll
        for (int j = 0; j < 4; ++j) {
            acc[0][j] = __builtin_amdgcn_wmma_f32_16x16x32_bf16(
                false, a0.v, false, bf[j].v, (short)0, acc[0][j], false, false);
            acc[1][j] = __builtin_amdgcn_wmma_f32_16x16x32_bf16(
                false, a1.v, false, bf[j].v, (short)0, acc[1][j], false, false);
        }
    }

    // C/D layout: VGPR r, lane l -> (M = half*8 + r, N = l&15)
    #pragma unroll
    for (int i = 0; i < 2; ++i) {
        #pragma unroll
        for (int j = 0; j < 4; ++j) {
            const int mBase = m0 + 16 * i + half * 8;
            const int nn    = n0 + 16 * j + l15;
            const float bv  = bias[nn];
            float ew = 0.0f;
            if (MODE == 0 || MODE == 1) ew = extraW[(size_t)nn * extraStride];
            #pragma unroll
            for (int r = 0; r < 8; ++r) {
                const int mm = mBase + r;
                float v = acc[i][j][r] + bv;
                if (MODE == 0) {                       // gate a: sigmoid * carry
                    v += surprise[mm >> 11] * ew;
                    v  = 1.0f / (1.0f + __expf(-v));
                    v *= carry[mm];
                    Cf[(size_t)mm * N + nn] = v;
                } else if (MODE == 1) {                // gate b: tanh
                    v += surprise[mm >> 11] * ew;
                    v  = tanhf(v);
                    Cf[(size_t)mm * N + nn] = v;
                } else if (MODE == 2) {                // W_o + residual x
                    v += residual[(size_t)mm * N + nn];
                    Cf[(size_t)mm * N + nn] = v;
                } else if (MODE == 3) {                // FFN1 + exact GELU -> bf16
                    v = 0.5f * v * (1.0f + erff(v * 0.70710678118654752f));
                    Cb[(size_t)mm * N + nn] = (bf16)v;
                } else {                               // FFN2 + residual out
                    v += residual[(size_t)mm * N + nn];
                    Cf[(size_t)mm * N + nn] = v;
                }
            }
        }
    }
}

// ---------------- chunked affine scan: h_t = a_t*h_{t-1} + b_t ----------------

__global__ void scan_pass1(const float* __restrict__ a, const float* __restrict__ b,
                           float* __restrict__ csA, float* __restrict__ csB) {
    int g = blockIdx.x * blockDim.x + threadIdx.x;   // SEQ*NCH threads
    int seq = g & (SEQ - 1);
    int ch  = g >> 13;
    int d   = seq & (D_H - 1);
    int bb  = seq >> 10;
    size_t base = ((size_t)bb * P + (size_t)ch * CLEN) * D_H + d;
    float Ap = 1.0f, Bc = 0.0f;
    for (int t = 0; t < CLEN; ++t) {
        float av = a[base + (size_t)t * D_H];
        float bv = b[base + (size_t)t * D_H];
        Bc = av * Bc + bv;
        Ap = av * Ap;
    }
    csA[(size_t)ch * SEQ + seq] = Ap;
    csB[(size_t)ch * SEQ + seq] = Bc;
}

__global__ void scan_pass2(const float* __restrict__ csA, const float* __restrict__ csB,
                           const float* __restrict__ h0, float* __restrict__ pre) {
    int seq = blockIdx.x * blockDim.x + threadIdx.x; // SEQ threads; seq = b*D_H + d
    float h = h0[seq];
    for (int c = 0; c < NCH; ++c) {
        pre[(size_t)c * SEQ + seq] = h;
        h = csA[(size_t)c * SEQ + seq] * h + csB[(size_t)c * SEQ + seq];
    }
}

// No __restrict__: hOut aliases aIn (in-place, each element read before written).
__global__ void scan_pass3(const float* aIn, const float* __restrict__ b,
                           const float* __restrict__ pre, float* hOut) {
    int g = blockIdx.x * blockDim.x + threadIdx.x;
    int seq = g & (SEQ - 1);
    int ch  = g >> 13;
    int d   = seq & (D_H - 1);
    int bb  = seq >> 10;
    size_t base = ((size_t)bb * P + (size_t)ch * CLEN) * D_H + d;
    float h = pre[(size_t)ch * SEQ + seq];
    for (int t = 0; t < CLEN; ++t) {
        size_t idx = base + (size_t)t * D_H;
        float av = aIn[idx];
        float bv = b[idx];
        h = av * h + bv;
        hOut[idx] = h;
    }
}

// ---------------- fused dual LayerNorm (out fp32 + z bf16) ----------------

__device__ __forceinline__ float block_sum(float v, float* red) {
    int t = threadIdx.x;
    red[t] = v;
    __syncthreads();
    for (int s = 128; s > 0; s >>= 1) {
        if (t < s) red[t] += red[t + s];
        __syncthreads();
    }
    float r = red[0];
    __syncthreads();
    return r;
}

__global__ __launch_bounds__(256) void ln_dual_kernel(
    const float* __restrict__ res, const float* __restrict__ g1, const float* __restrict__ b1,
    const float* __restrict__ g2, const float* __restrict__ b2,
    float* __restrict__ outb, bf16* __restrict__ zb)
{
    __shared__ float red[256];
    const int row = blockIdx.x;
    const int t   = threadIdx.x;
    const float* r = res + (size_t)row * D_H;

    float v[4];
    float s = 0.0f;
    #pragma unroll
    for (int i = 0; i < 4; ++i) { v[i] = r[t + 256 * i]; s += v[i]; }
    s = block_sum(s, red);
    const float mean = s * (1.0f / D_H);
    float vs = 0.0f;
    #pragma unroll
    for (int i = 0; i < 4; ++i) { float d = v[i] - mean; vs += d * d; }
    vs = block_sum(vs, red);
    const float rstd = rsqrtf(vs * (1.0f / D_H) + 1e-5f);

    float o[4];
    float s2 = 0.0f;
    #pragma unroll
    for (int i = 0; i < 4; ++i) {
        int idx = t + 256 * i;
        o[i] = (v[i] - mean) * rstd * g1[idx] + b1[idx];
        outb[(size_t)row * D_H + idx] = o[i];
        s2 += o[i];
    }
    s2 = block_sum(s2, red);
    const float mean2 = s2 * (1.0f / D_H);
    float vs2 = 0.0f;
    #pragma unroll
    for (int i = 0; i < 4; ++i) { float d = o[i] - mean2; vs2 += d * d; }
    vs2 = block_sum(vs2, red);
    const float rstd2 = rsqrtf(vs2 * (1.0f / D_H) + 1e-5f);
    #pragma unroll
    for (int i = 0; i < 4; ++i) {
        int idx = t + 256 * i;
        float z = (o[i] - mean2) * rstd2 * g2[idx] + b2[idx];
        zb[(size_t)row * D_H + idx] = (bf16)z;
    }
}

// ---------------- orchestration ----------------

extern "C" void kernel_launch(void* const* d_in, const int* in_sizes, int n_in,
                              void* d_out, int out_size, void* d_ws, size_t ws_size,
                              hipStream_t stream) {
    const float* x        = (const float*)d_in[0];
    const float* ypm      = (const float*)d_in[1];
    const float* ywm      = (const float*)d_in[2];
    const float* yem      = (const float*)d_in[3];
    const float* surprise = (const float*)d_in[4];
    const float* carry    = (const float*)d_in[5];
    const float* h0       = (const float*)d_in[6];
    const float* gate_a_w = (const float*)d_in[7];
    const float* gate_a_b = (const float*)d_in[8];
    const float* gate_b_w = (const float*)d_in[9];
    const float* gate_b_b = (const float*)d_in[10];
    const float* W_o_w    = (const float*)d_in[11];
    const float* W_o_b    = (const float*)d_in[12];
    const float* norm_g   = (const float*)d_in[13];
    const float* norm_b   = (const float*)d_in[14];
    const float* ffn_g    = (const float*)d_in[15];
    const float* ffn_b    = (const float*)d_in[16];
    const float* ffn_w1   = (const float*)d_in[17];
    const float* ffn_b1   = (const float*)d_in[18];
    const float* ffn_w2   = (const float*)d_in[19];
    const float* ffn_b2   = (const float*)d_in[20];

    char* ws = (char*)d_ws;
    bf16*  U    = (bf16*)(ws + OFF_U);
    bf16*  WA   = (bf16*)(ws + OFF_WA);
    bf16*  WB   = (bf16*)(ws + OFF_WB);
    bf16*  WO   = (bf16*)(ws + OFF_WO);
    bf16*  W1   = (bf16*)(ws + OFF_W1);
    bf16*  W2   = (bf16*)(ws + OFF_W2);
    float* Abuf = (float*)(ws + OFF_ABUF);   // a_eff -> h -> out
    float* Bbuf = (float*)(ws + OFF_BBUF);   // b -> res1
    bf16*  Zbf  = (bf16*)(ws + OFF_Z);       // h_bf16 -> z_bf16
    float* CSA  = (float*)(ws + OFF_CSA);
    float* CSB  = (float*)(ws + OFF_CSB);
    float* PRE  = (float*)(ws + OFF_PRE);
    bf16*  ZF   = U;                          // FFN1 output reuses u's region

    // 1) pack + convert
    pack_u_kernel<<<4096, 256, 0, stream>>>(x, ypm, ywm, yem, U);
    cvt_f32_bf16_kernel<<<2048, 256, 0, stream>>>(gate_a_w, WA, 1024, 4097, 4096);
    cvt_f32_bf16_kernel<<<2048, 256, 0, stream>>>(gate_b_w, WB, 1024, 4097, 4096);
    cvt_f32_bf16_kernel<<<1024, 256, 0, stream>>>(W_o_w,   WO, 1024, 1024, 1024);
    cvt_f32_bf16_kernel<<<2048, 256, 0, stream>>>(ffn_w1,  W1, 4096, 1024, 1024);
    cvt_f32_bf16_kernel<<<2048, 256, 0, stream>>>(ffn_w2,  W2, 1024, 4096, 4096);

    // 2) gate GEMMs (fused surprise column, sigmoid*carry / tanh)
    dim3 gGate(D_H / 128, MROWS / 128);
    gemm_bf16_wmma<0><<<gGate, 256, 0, stream>>>(U, WA, Abuf, nullptr, MROWS, D_H, 4096,
        gate_a_b, gate_a_w + 4096, 4097, surprise, carry, nullptr);
    gemm_bf16_wmma<1><<<gGate, 256, 0, stream>>>(U, WB, Bbuf, nullptr, MROWS, D_H, 4096,
        gate_b_b, gate_b_w + 4096, 4097, surprise, nullptr, nullptr);

    // 3) chunked affine scan (h overwrites a_eff buffer)
    scan_pass1<<<(SEQ * NCH) / 256, 256, 0, stream>>>(Abuf, Bbuf, CSA, CSB);
    scan_pass2<<<SEQ / 256, 256, 0, stream>>>(CSA, CSB, h0, PRE);
    scan_pass3<<<(SEQ * NCH) / 256, 256, 0, stream>>>(Abuf, Bbuf, PRE, Abuf);

    // 4) h -> bf16, W_o GEMM (+x residual) -> res1 (overwrites b buffer)
    cvt_f32_bf16_kernel<<<2048, 256, 0, stream>>>(Abuf, Zbf, MROWS, 1024, 1024);
    dim3 gWo(D_H / 128, MROWS / 128);
    gemm_bf16_wmma<2><<<gWo, 256, 0, stream>>>(Zbf, WO, Bbuf, nullptr, MROWS, D_H, 1024,
        W_o_b, nullptr, 0, nullptr, nullptr, x);

    // 5) dual LayerNorm: out (fp32, overwrites h buffer) + z (bf16, overwrites h_bf16)
    ln_dual_kernel<<<MROWS, 256, 0, stream>>>(Bbuf, norm_g, norm_b, ffn_g, ffn_b, Abuf, Zbf);

    // 6) FFN
    dim3 gF1(D_FF / 128, MROWS / 128);
    gemm_bf16_wmma<3><<<gF1, 256, 0, stream>>>(Zbf, W1, nullptr, ZF, MROWS, D_FF, 1024,
        ffn_b1, nullptr, 0, nullptr, nullptr, nullptr);
    dim3 gF2(D_H / 128, MROWS / 128);
    gemm_bf16_wmma<4><<<gF2, 256, 0, stream>>>(ZF, W2, (float*)d_out, nullptr, MROWS, D_H, 4096,
        ffn_b2, nullptr, 0, nullptr, nullptr, Abuf);
}
